// FormulaNetSAT_77403900609207
// MI455X (gfx1250) — matrix operand order repaired and checked by reading, not compiled, not run on hardware
//
#include <hip/hip_runtime.h>
#include <hip/hip_bf16.h>
#include <stdint.h>
#include <stddef.h>

#define EMB   128
#define EDGEF 32
#define KCAT  (2 * EMB + EDGEF)   // 288
#define NK1   (KCAT / 32)         // 9 K-slices in GEMM1
#define ITERS 2
#define EPSV  1e-5f

#ifndef __has_builtin
#define __has_builtin(x) 0
#endif

#if __has_builtin(__builtin_amdgcn_global_load_async_to_lds_b128)
#define HAVE_ASYNC_LDS 1
#else
#define HAVE_ASYNC_LDS 0
#endif

typedef __attribute__((ext_vector_type(16))) __bf16 v16bf;
typedef __attribute__((ext_vector_type(8)))  __bf16 v8bf;
typedef __attribute__((ext_vector_type(8)))  float  v8f;
typedef __attribute__((ext_vector_type(4)))  float  v4f_;
typedef __attribute__((ext_vector_type(4)))  int    v4i_;

// 16-byte global -> LDS copy: async (ASYNCcnt-tracked, no VGPR round trip).
static __device__ __forceinline__ void copy16_g2l(__bf16* l, const __bf16* g) {
#if HAVE_ASYNC_LDS
  __builtin_amdgcn_global_load_async_to_lds_b128(
      (__attribute__((address_space(1))) v4i_*)(void*)g,
      (__attribute__((address_space(3))) v4i_*)(void*)l,
      0, 0);
#else
  *(uint4*)l = *(const uint4*)g;
#endif
}

static __device__ __forceinline__ void async_wait_all() {
#if HAVE_ASYNC_LDS
#if __has_builtin(__builtin_amdgcn_s_wait_asynccnt)
  __builtin_amdgcn_s_wait_asynccnt(0);
#else
  asm volatile("s_wait_asynccnt 0" ::: "memory");
#endif
#endif
}

static __device__ __forceinline__ v16bf frag_from2(const __bf16* p0, const __bf16* p1) {
  v8bf lo = *(const v8bf*)p0;
  v8bf hi = *(const v8bf*)p1;
  v16bf r;
#pragma unroll
  for (int i = 0; i < 8; ++i) { r[i] = lo[i]; r[i + 8] = hi[i]; }
  return r;
}

// ---------------------------------------------------------------------------
// GEMM1 (edge MLP layer 1): A = gathered [x_i | x_j | edge_attr] (64 x 288 bf16),
// B = W1t (pre-transposed, 128 x 288 N-major bf16), double-buffered K-slices,
// fused +b1 + BN-stat atomics; H written non-temporal (one-shot 205 MB stream).
// ---------------------------------------------------------------------------
__global__ __launch_bounds__(256)
void edge_gemm1_kernel(const __bf16* __restrict__ nodesB,
                       const int* __restrict__ idxI,
                       const int* __restrict__ idxJ,
                       const float* __restrict__ eattr,
                       const __bf16* __restrict__ Wt,   // EMB x KCAT (n-major)
                       const float* __restrict__ bvec,
                       float* __restrict__ H,
                       float* __restrict__ gsum,
                       float* __restrict__ gsq,
                       int E)
{
  __shared__ __align__(16) __bf16 sA[64 * KCAT];       // 36.9 KB
  __shared__ __align__(16) __bf16 sBd[2][32 * EMB];    // 2 x 8.2 KB double buffer
  __shared__ int sI[64], sJ[64];

  const int tid  = threadIdx.x;
  const int base = blockIdx.x * 64;
  const bool fullTile = (base + 64 <= E);

  if (tid < 64) {
    const int e = base + tid;
    sI[tid] = (e < E) ? idxI[e] : 0;
    sJ[tid] = (e < E) ? idxJ[e] : 0;
  }
  __syncthreads();

  if (fullTile) {
    // async gather: per-lane 16B chunks straight into LDS
    for (int q = tid; q < 64 * 16; q += 256) {
      const int r = q >> 4, ch = q & 15;
      copy16_g2l(sA + r * KCAT + ch * 8, nodesB + (size_t)sI[r] * EMB + ch * 8);
    }
    for (int q = tid; q < 64 * 16; q += 256) {
      const int r = q >> 4, ch = q & 15;
      copy16_g2l(sA + r * KCAT + EMB + ch * 8, nodesB + (size_t)sJ[r] * EMB + ch * 8);
    }
  } else {
    for (int q = tid; q < 64 * 16; q += 256) {
      const int r = q >> 4, ch = q & 15;
      uint4 v = make_uint4(0u, 0u, 0u, 0u);
      if (base + r < E) v = ((const uint4*)(nodesB + (size_t)sI[r] * EMB))[ch];
      ((uint4*)(sA + r * KCAT))[ch] = v;
    }
    for (int q = tid; q < 64 * 16; q += 256) {
      const int r = q >> 4, ch = q & 15;
      uint4 v = make_uint4(0u, 0u, 0u, 0u);
      if (base + r < E) v = ((const uint4*)(nodesB + (size_t)sJ[r] * EMB))[ch];
      ((uint4*)(sA + r * KCAT + EMB))[ch] = v;
    }
  }
  // edge_attr fp32 -> bf16 (VALU convert, packed 8B stores)
  for (int q = tid; q < 64 * 8; q += 256) {
    const int r = q >> 3, ch = q & 7;
    float4 f = make_float4(0.f, 0.f, 0.f, 0.f);
    if (base + r < E) f = ((const float4*)(eattr + (size_t)(base + r) * EDGEF))[ch];
    union { __bf16 h[4]; uint2 u; } pk;
    pk.h[0] = (__bf16)f.x; pk.h[1] = (__bf16)f.y;
    pk.h[2] = (__bf16)f.z; pk.h[3] = (__bf16)f.w;
    *(uint2*)(sA + r * KCAT + 2 * EMB + ch * 4) = pk.u;
  }
  // prologue: stage W slice 0 into buffer 0 (concurrent with gathers)
  for (int q = tid; q < 128 * 4; q += 256) {
    const int n = q >> 2, part = q & 3;
    copy16_g2l(sBd[0] + n * 32 + part * 8, Wt + (size_t)n * KCAT + part * 8);
  }
  async_wait_all();
  __syncthreads();

  const int w    = tid >> 5;
  const int lane = tid & 31;
  const int wm   = w & 3;            // which 16-row tile
  const int wn0  = (w >> 2) * 64;    // 64-wide column strip
  const int half = lane >> 4;
  const int lm   = lane & 15;

  v8f acc[4] = {};

  for (int kk = 0; kk < NK1; ++kk) {
    const int cur = kk & 1;
    // issue async loads of the NEXT slice while computing on the current one
    if (kk + 1 < NK1) {
      for (int q = tid; q < 128 * 4; q += 256) {
        const int n = q >> 2, part = q & 3;
        copy16_g2l(sBd[cur ^ 1] + n * 32 + part * 8,
                   Wt + (size_t)n * KCAT + (kk + 1) * 32 + part * 8);
      }
    }

    const __bf16* ap = sA + (wm * 16 + lm) * KCAT + kk * 32 + half * 16;
    const v16bf a = frag_from2(ap, ap + 8);
#pragma unroll
    for (int nt = 0; nt < 4; ++nt) {
      const v16bf b = *(const v16bf*)(sBd[cur] + (wn0 + nt * 16 + lm) * 32 + half * 16);
      acc[nt] = __builtin_amdgcn_wmma_f32_16x16x32_bf16(
          false, a, false, b, (short)0, acc[nt], false, false);
    }

    async_wait_all();     // next slice landed
    __syncthreads();      // everyone done reading cur + next slice visible
  }

#pragma unroll
  for (int nt = 0; nt < 4; ++nt) {
    const int col = wn0 + nt * 16 + lm;
    const float bb = bvec[col];
    float s = 0.f, sq = 0.f;
#pragma unroll
    for (int v = 0; v < 8; ++v) {
      const int gr = base + wm * 16 + half * 8 + v;  // M = v + 8*half (ISA layout)
      const float c = acc[nt][v] + bb;
      if (gr < E) {
        __builtin_nontemporal_store(c, H + (size_t)gr * EMB + col);
        s += c; sq += c * c;
      }
    }
    s  += __shfl_xor(s, 16);
    sq += __shfl_xor(sq, 16);
    if (half == 0) {
      atomicAdd(&gsum[col], s);
      atomicAdd(&gsq[col],  sq);
    }
  }
}

// ---------------------------------------------------------------------------
// GEMM2 (edge MLP layer 2): input = BN1+ReLU(h1) on the fly -> bf16 LDS,
// B = W2t (128 x 128 n-major), output IN PLACE over H (NT load + NT store),
// fused +b2 + BN2 stats.
// ---------------------------------------------------------------------------
__global__ __launch_bounds__(256)
void edge_gemm2_kernel(float* __restrict__ H,
                       const float* __restrict__ scale1,
                       const float* __restrict__ bias1,
                       const __bf16* __restrict__ Wt,
                       const float* __restrict__ bvec,
                       float* __restrict__ gsum,
                       float* __restrict__ gsq,
                       int E)
{
  __shared__ __align__(16) __bf16 sA[64 * EMB];    // 16 KB
  __shared__ __align__(16) __bf16 sB[EMB * EMB];   // 32 KB (n-major)

  const int tid  = threadIdx.x;
  const int base = blockIdx.x * 64;

  // whole W2t panel, fully contiguous 16B chunks (async)
  for (int q = tid; q < EMB * EMB / 8; q += 256)
    copy16_g2l(sB + q * 8, Wt + q * 8);

  for (int q = tid; q < 64 * 32; q += 256) {       // 64 rows x 32 float4
    const int r = q >> 5, ch = q & 31;
    v4f_ f = {0.f, 0.f, 0.f, 0.f};
    if (base + r < E)
      f = __builtin_nontemporal_load((const v4f_*)(H + (size_t)(base + r) * EMB) + ch);
    const v4f_ sc = ((const v4f_*)scale1)[ch];
    const v4f_ bi = ((const v4f_*)bias1)[ch];
    union { __bf16 h[4]; uint2 u; } pk;
    pk.h[0] = (__bf16)fmaxf(f[0] * sc[0] + bi[0], 0.f);
    pk.h[1] = (__bf16)fmaxf(f[1] * sc[1] + bi[1], 0.f);
    pk.h[2] = (__bf16)fmaxf(f[2] * sc[2] + bi[2], 0.f);
    pk.h[3] = (__bf16)fmaxf(f[3] * sc[3] + bi[3], 0.f);
    *(uint2*)(sA + r * EMB + ch * 4) = pk.u;
  }
  async_wait_all();
  __syncthreads();

  const int w    = tid >> 5;
  const int lane = tid & 31;
  const int wm   = w & 3;
  const int wn0  = (w >> 2) * 64;
  const int half = lane >> 4;
  const int lm   = lane & 15;

  v8f acc[4] = {};
#pragma unroll
  for (int kk = 0; kk < EMB / 32; ++kk) {
    const __bf16* ap = sA + (wm * 16 + lm) * EMB + kk * 32 + half * 16;
    const v16bf a = frag_from2(ap, ap + 8);
#pragma unroll
    for (int nt = 0; nt < 4; ++nt) {
      const v16bf b = *(const v16bf*)(sB + (wn0 + nt * 16 + lm) * EMB + kk * 32 + half * 16);
      acc[nt] = __builtin_amdgcn_wmma_f32_16x16x32_bf16(
          false, a, false, b, (short)0, acc[nt], false, false);
    }
  }

#pragma unroll
  for (int nt = 0; nt < 4; ++nt) {
    const int col = wn0 + nt * 16 + lm;
    const float bb = bvec[col];
    float s = 0.f, sq = 0.f;
#pragma unroll
    for (int v = 0; v < 8; ++v) {
      const int gr = base + wm * 16 + half * 8 + v;
      const float c = acc[nt][v] + bb;
      if (gr < E) {
        __builtin_nontemporal_store(c, H + (size_t)gr * EMB + col);
        s += c; sq += c * c;
      }
    }
    s  += __shfl_xor(s, 16);
    sq += __shfl_xor(sq, 16);
    if (half == 0) {
      atomicAdd(&gsum[col], s);
      atomicAdd(&gsq[col],  sq);
    }
  }
}

// ---------------------------------------------------------------------------
// Node GEMM: A = z (bf16, precomputed), B = fWt (n-major), out U + b + stats.
// ---------------------------------------------------------------------------
__global__ __launch_bounds__(256)
void rows_gemm_kernel(const __bf16* __restrict__ Abf,
                      const __bf16* __restrict__ Wt,
                      const float* __restrict__ bvec,
                      float* __restrict__ U,
                      float* __restrict__ gsum,
                      float* __restrict__ gsq,
                      int R)
{
  __shared__ __align__(16) __bf16 sA[64 * EMB];
  __shared__ __align__(16) __bf16 sB[EMB * EMB];

  const int tid  = threadIdx.x;
  const int base = blockIdx.x * 64;

  for (int q = tid; q < EMB * EMB / 8; q += 256)
    copy16_g2l(sB + q * 8, Wt + q * 8);

  if (base + 64 <= R) {
    for (int q = tid; q < 64 * 16; q += 256) {
      const int r = q >> 4, ch = q & 15;
      copy16_g2l(sA + r * EMB + ch * 8, Abf + (size_t)(base + r) * EMB + ch * 8);
    }
  } else {
    for (int q = tid; q < 64 * 16; q += 256) {
      const int r = q >> 4, ch = q & 15;
      uint4 v = make_uint4(0u, 0u, 0u, 0u);
      if (base + r < R) v = ((const uint4*)(Abf + (size_t)(base + r) * EMB))[ch];
      ((uint4*)(sA + r * EMB))[ch] = v;
    }
  }
  async_wait_all();
  __syncthreads();

  const int w    = tid >> 5;
  const int lane = tid & 31;
  const int wm   = w & 3;
  const int wn0  = (w >> 2) * 64;
  const int half = lane >> 4;
  const int lm   = lane & 15;

  v8f acc[4] = {};
#pragma unroll
  for (int kk = 0; kk < EMB / 32; ++kk) {
    const __bf16* ap = sA + (wm * 16 + lm) * EMB + kk * 32 + half * 16;
    const v16bf a = frag_from2(ap, ap + 8);
#pragma unroll
    for (int nt = 0; nt < 4; ++nt) {
      const v16bf b = *(const v16bf*)(sB + (wn0 + nt * 16 + lm) * EMB + kk * 32 + half * 16);
      acc[nt] = __builtin_amdgcn_wmma_f32_16x16x32_bf16(
          false, a, false, b, (short)0, acc[nt], false, false);
    }
  }

#pragma unroll
  for (int nt = 0; nt < 4; ++nt) {
    const int col = wn0 + nt * 16 + lm;
    const float bb = bvec[col];
    float s = 0.f, sq = 0.f;
#pragma unroll
    for (int v = 0; v < 8; ++v) {
      const int gr = base + wm * 16 + half * 8 + v;
      const float c = acc[nt][v] + bb;
      if (gr < R) {
        U[(size_t)gr * EMB + col] = c;
        s += c; sq += c * c;
      }
    }
    s  += __shfl_xor(s, 16);
    sq += __shfl_xor(sq, 16);
    if (half == 0) {
      atomicAdd(&gsum[col], s);
      atomicAdd(&gsq[col],  sq);
    }
  }
}

// ---------------------------------------------------------------------------
// Small helper kernels
// ---------------------------------------------------------------------------
__global__ void bn_finalize_kernel(const float* __restrict__ gsum,
                                   const float* __restrict__ gsq,
                                   const float* __restrict__ g,
                                   const float* __restrict__ be,
                                   float* __restrict__ scale,
                                   float* __restrict__ bias,
                                   float invR)
{
  const int c = threadIdx.x;
  const float m   = gsum[c] * invR;
  const float var = gsq[c] * invR - m * m;
  const float sc  = g[c] * rsqrtf(var + EPSV);
  scale[c] = sc;
  bias[c]  = be[c] - m * sc;
}

__global__ void scatter_kernel(const float* __restrict__ H,
                               const float* __restrict__ scale,
                               const float* __restrict__ bias,
                               const int* __restrict__ idxI,
                               float* __restrict__ agg,
                               long long total)
{
  const long long gid = (long long)blockIdx.x * 256 + threadIdx.x;
  if (gid >= total) return;
  const int c = (int)(gid & (EMB - 1));
  const int e = (int)(gid >> 7);
  const float hv = __builtin_nontemporal_load(H + gid);
  const float v = fmaxf(hv * scale[c] + bias[c], 0.f);
  atomicAdd(&agg[(size_t)idxI[e] * EMB + c], v);
}

__global__ void deg_kernel(const int* __restrict__ idx, float* __restrict__ deg, int E) {
  const int e = blockIdx.x * 256 + threadIdx.x;
  if (e < E) atomicAdd(&deg[idx[e]], 1.0f);
}

__global__ void deginv_kernel(float* __restrict__ deg, int N) {
  const int i = blockIdx.x * 256 + threadIdx.x;
  if (i < N) { const float d = deg[i]; deg[i] = (d > 0.f) ? (1.0f / d) : 0.f; }
}

__global__ void build_z_kernel(const float* __restrict__ nodes,
                               const float* __restrict__ aggP,
                               const float* __restrict__ aggC,
                               const float* __restrict__ dinvP,
                               const float* __restrict__ dinvC,
                               __bf16* __restrict__ zb,
                               long long total)
{
  const long long gid = (long long)blockIdx.x * 256 + threadIdx.x;
  if (gid >= total) return;
  const int row = (int)(gid >> 7);
  const float z = nodes[gid] + aggP[gid] * dinvP[row] + aggC[gid] * dinvC[row];
  zb[gid] = (__bf16)z;
}

__global__ void update_kernel(float* __restrict__ nodes,
                              __bf16* __restrict__ nodesB,
                              const float* __restrict__ U,
                              const float* __restrict__ scale,
                              const float* __restrict__ bias,
                              float* __restrict__ outF,
                              int writeOut,
                              long long total)
{
  const long long gid = (long long)blockIdx.x * 256 + threadIdx.x;
  if (gid >= total) return;
  const int c = (int)(gid & (EMB - 1));
  const float u  = fmaxf(U[gid] * scale[c] + bias[c], 0.f);
  const float nv = nodes[gid] + u;
  nodes[gid]  = nv;
  nodesB[gid] = (__bf16)nv;
  if (writeOut) outF[gid] = nv;
}

// fp32 (K x N row-major) -> bf16 transposed (N x K, n-major)
__global__ void f2bfT_kernel(const float* __restrict__ s, __bf16* __restrict__ d,
                             int K, int Ncols)
{
  const int idx = blockIdx.x * 256 + threadIdx.x;
  if (idx >= K * Ncols) return;
  const int k = idx / Ncols, n = idx % Ncols;
  d[(size_t)n * K + k] = (__bf16)s[idx];
}

__global__ void init_nodes_kernel(const float* __restrict__ s,
                                  float* __restrict__ nf,
                                  __bf16* __restrict__ nb,
                                  long long total)
{
  const long long gid = (long long)blockIdx.x * 256 + threadIdx.x;
  if (gid >= total) return;
  const float v = s[gid];
  nf[gid] = v;
  nb[gid] = (__bf16)v;
}

// ---------------------------------------------------------------------------
// Host orchestration
// ---------------------------------------------------------------------------
extern "C" void kernel_launch(void* const* d_in, const int* in_sizes, int n_in,
                              void* d_out, int out_size, void* d_ws, size_t ws_size,
                              hipStream_t stream)
{
  (void)n_in; (void)out_size; (void)ws_size;

  const float* nodes0 = (const float*)d_in[0];
  const int*   edges  = (const int*)d_in[1];
  const float* eattr  = (const float*)d_in[2];
  const float* fW   = (const float*)d_in[3];
  const float* fb   = (const float*)d_in[4];
  const float* fg   = (const float*)d_in[5];
  const float* fbe  = (const float*)d_in[6];
  const float* pW1  = (const float*)d_in[7];
  const float* pb1  = (const float*)d_in[8];
  const float* pg1  = (const float*)d_in[9];
  const float* pbe1 = (const float*)d_in[10];
  const float* pW2  = (const float*)d_in[11];
  const float* pb2  = (const float*)d_in[12];
  const float* pg2  = (const float*)d_in[13];
  const float* pbe2 = (const float*)d_in[14];
  const float* cW1  = (const float*)d_in[15];
  const float* cb1  = (const float*)d_in[16];
  const float* cg1  = (const float*)d_in[17];
  const float* cbe1 = (const float*)d_in[18];
  const float* cW2  = (const float*)d_in[19];
  const float* cb2  = (const float*)d_in[20];
  const float* cg2  = (const float*)d_in[21];
  const float* cbe2 = (const float*)d_in[22];

  const int N = in_sizes[0] / EMB;
  const int E = in_sizes[1] / 2;
  const int* srcIdx = edges;       // edges[0]
  const int* dstIdx = edges + E;   // edges[1]

  char* wsp = (char*)d_ws;
  size_t off = 0;
  auto alloc = [&](size_t bytes) -> void* {
    off = (off + 255) & ~(size_t)255;
    void* p = wsp + off;
    off += bytes;
    return p;
  };

  const size_t totN = (size_t)N * EMB;
  const size_t totE = (size_t)E * EMB;

  float*  nodesF = (float*)alloc(totN * 4);
  __bf16* nodesB = (__bf16*)alloc(totN * 2);
  __bf16* zB     = (__bf16*)alloc(totN * 2);
  float*  U      = (float*)alloc(totN * 4);
  float*  aggP   = (float*)alloc(totN * 4);
  float*  aggC   = (float*)alloc(totN * 4);
  float*  dinvP  = (float*)alloc((size_t)N * 4);
  float*  dinvC  = (float*)alloc((size_t)N * 4);
  float*  H      = (float*)alloc(totE * 4);
  __bf16* pW1t   = (__bf16*)alloc((size_t)KCAT * EMB * 2);
  __bf16* cW1t   = (__bf16*)alloc((size_t)KCAT * EMB * 2);
  __bf16* pW2t   = (__bf16*)alloc((size_t)EMB * EMB * 2);
  __bf16* cW2t   = (__bf16*)alloc((size_t)EMB * EMB * 2);
  __bf16* fWt    = (__bf16*)alloc((size_t)EMB * EMB * 2);
  float*  stSum  = (float*)alloc(128 * 4);
  float*  stSq   = (float*)alloc(128 * 4);
  float*  stScl  = (float*)alloc(128 * 4);
  float*  stBia  = (float*)alloc(128 * 4);

  // one-time transposed bf16 weight panels (N-major => contiguous LDS staging)
  f2bfT_kernel<<<(KCAT * EMB + 255) / 256, 256, 0, stream>>>(pW1, pW1t, KCAT, EMB);
  f2bfT_kernel<<<(KCAT * EMB + 255) / 256, 256, 0, stream>>>(cW1, cW1t, KCAT, EMB);
  f2bfT_kernel<<<(EMB * EMB + 255) / 256, 256, 0, stream>>>(pW2, pW2t, EMB, EMB);
  f2bfT_kernel<<<(EMB * EMB + 255) / 256, 256, 0, stream>>>(cW2, cW2t, EMB, EMB);
  f2bfT_kernel<<<(EMB * EMB + 255) / 256, 256, 0, stream>>>(fW,  fWt,  EMB, EMB);

  const int NBt = (int)((totN + 255) / 256);
  init_nodes_kernel<<<NBt, 256, 0, stream>>>(nodes0, nodesF, nodesB, (long long)totN);

  // degree (computed once): parent aggregates at dst, child at src
  (void)hipMemsetAsync(dinvP, 0, (size_t)N * 4, stream);
  (void)hipMemsetAsync(dinvC, 0, (size_t)N * 4, stream);
  deg_kernel<<<(E + 255) / 256, 256, 0, stream>>>(dstIdx, dinvP, E);
  deg_kernel<<<(E + 255) / 256, 256, 0, stream>>>(srcIdx, dinvC, E);
  deginv_kernel<<<(N + 255) / 256, 256, 0, stream>>>(dinvP, N);
  deginv_kernel<<<(N + 255) / 256, 256, 0, stream>>>(dinvC, N);

  const int EB = (E + 63) / 64;
  const int RB = (N + 63) / 64;
  const int SB = (int)((totE + 255) / 256);

  for (int it = 0; it < ITERS; ++it) {
    (void)hipMemsetAsync(aggP, 0, totN * 4, stream);
    (void)hipMemsetAsync(aggC, 0, totN * 4, stream);

    // ---- parent direction: idx_i = dst, idx_j = src ----
    (void)hipMemsetAsync(stSum, 0, 512, stream);
    (void)hipMemsetAsync(stSq,  0, 512, stream);
    edge_gemm1_kernel<<<EB, 256, 0, stream>>>(nodesB, dstIdx, srcIdx, eattr,
                                              pW1t, pb1, H, stSum, stSq, E);
    bn_finalize_kernel<<<1, 128, 0, stream>>>(stSum, stSq, pg1, pbe1, stScl, stBia, 1.0f / E);
    (void)hipMemsetAsync(stSum, 0, 512, stream);
    (void)hipMemsetAsync(stSq,  0, 512, stream);
    edge_gemm2_kernel<<<EB, 256, 0, stream>>>(H, stScl, stBia, pW2t, pb2, stSum, stSq, E);
    bn_finalize_kernel<<<1, 128, 0, stream>>>(stSum, stSq, pg2, pbe2, stScl, stBia, 1.0f / E);
    scatter_kernel<<<SB, 256, 0, stream>>>(H, stScl, stBia, dstIdx, aggP, (long long)totE);

    // ---- child direction: idx_i = src, idx_j = dst ----
    (void)hipMemsetAsync(stSum, 0, 512, stream);
    (void)hipMemsetAsync(stSq,  0, 512, stream);
    edge_gemm1_kernel<<<EB, 256, 0, stream>>>(nodesB, srcIdx, dstIdx, eattr,
                                              cW1t, cb1, H, stSum, stSq, E);
    bn_finalize_kernel<<<1, 128, 0, stream>>>(stSum, stSq, cg1, cbe1, stScl, stBia, 1.0f / E);
    (void)hipMemsetAsync(stSum, 0, 512, stream);
    (void)hipMemsetAsync(stSq,  0, 512, stream);
    edge_gemm2_kernel<<<EB, 256, 0, stream>>>(H, stScl, stBia, cW2t, cb2, stSum, stSq, E);
    bn_finalize_kernel<<<1, 128, 0, stream>>>(stSum, stSq, cg2, cbe2, stScl, stBia, 1.0f / E);
    scatter_kernel<<<SB, 256, 0, stream>>>(H, stScl, stBia, srcIdx, aggC, (long long)totE);

    // ---- node update ----
    build_z_kernel<<<NBt, 256, 0, stream>>>(nodesF, aggP, aggC, dinvP, dinvC, zB, (long long)totN);
    (void)hipMemsetAsync(stSum, 0, 512, stream);
    (void)hipMemsetAsync(stSq,  0, 512, stream);
    rows_gemm_kernel<<<RB, 256, 0, stream>>>(zB, fWt, fb, U, stSum, stSq, N);
    bn_finalize_kernel<<<1, 128, 0, stream>>>(stSum, stSq, fg, fbe, stScl, stBia, 1.0f / N);
    update_kernel<<<NBt, 256, 0, stream>>>(nodesF, nodesB, U, stScl, stBia,
                                           (float*)d_out, (it == ITERS - 1) ? 1 : 0,
                                           (long long)totN);
  }
}
